// SCOT_56856777064697
// MI455X (gfx1250) — matrix operand order repaired and checked by reading, not compiled, not run on hardware
//
#include <hip/hip_runtime.h>
#include <hip/hip_bf16.h>
#include <math.h>

typedef __attribute__((ext_vector_type(16))) _Float16 v16h;
typedef __attribute__((ext_vector_type(8)))  float    v8f;

#define C_CELLS 128
#define G_GENES 2000
#define CE_DIM  64
#define GE_DIM  64
#define H_DIM   256

__device__ __forceinline__ float eluf(float x) {
    return x > 0.f ? x : (__expf(x) - 1.f);
}

// ---- P1: ce = elu(ctrl @ ce_w1 + b1) @ ce_w2 + b2   (one block per cell) ----
__global__ void __launch_bounds__(H_DIM)
prep_ce_kernel(const float* __restrict__ ctrl,
               const float* __restrict__ ce_w1, const float* __restrict__ ce_b1,
               const float* __restrict__ ce_w2, const float* __restrict__ ce_b2,
               float* __restrict__ ce)
{
    __shared__ float hid[H_DIM];
    const int c = blockIdx.x;
    const int t = threadIdx.x;
    float acc = ce_b1[t];
    const float* crow = ctrl + c * G_GENES;
    for (int j = 0; j < G_GENES; ++j)
        acc += crow[j] * ce_w1[j * H_DIM + t];
    hid[t] = eluf(acc);
    __syncthreads();
    if (t < CE_DIM) {
        float a2 = ce_b2[t];
        for (int k = 0; k < H_DIM; ++k)
            a2 += hid[k] * ce_w2[k * CE_DIM + t];
        ce[c * CE_DIM + t] = a2;
    }
}

// ---- P2: A[c,h] = ce[c,:] @ g_w1[1:65,:] ----
__global__ void __launch_bounds__(H_DIM)
prep_A_kernel(const float* __restrict__ ce, const float* __restrict__ g_w1,
              float* __restrict__ A)
{
    const int c = blockIdx.x, h = threadIdx.x;
    float acc = 0.f;
    for (int e = 0; e < CE_DIM; ++e)
        acc += ce[c * CE_DIM + e] * g_w1[(1 + e) * H_DIM + h];
    A[c * H_DIM + h] = acc;
}

// ---- P3: B[g,h] = ge@g_w1[65:129] + shift*w129 + 128*w130 + b1 ----
__global__ void __launch_bounds__(H_DIM)
prep_B_kernel(const int* __restrict__ gidxs, const float* __restrict__ shiftv,
              const float* __restrict__ gtab, const float* __restrict__ g_w1,
              const float* __restrict__ g_b1, float* __restrict__ B)
{
    const int g = blockIdx.x, h = threadIdx.x;
    const int gi = gidxs[g];
    float acc = g_b1[h]
              + shiftv[gi] * g_w1[129 * H_DIM + h]
              + (float)C_CELLS * g_w1[130 * H_DIM + h];
    for (int e = 0; e < GE_DIM; ++e)
        acc += gtab[gi * GE_DIM + e] * g_w1[(65 + e) * H_DIM + h];
    B[g * H_DIM + h] = acc;
}

// ---- P4: W2aT[n,k] = (f16) g_w2[k,n]  (transpose first 256 rows of g_w2) ----
__global__ void __launch_bounds__(H_DIM)
prep_w2t_kernel(const float* __restrict__ g_w2, _Float16* __restrict__ W2aT)
{
    const int idx = blockIdx.x * H_DIM + threadIdx.x;  // 0..65535
    const int n = idx >> 8, k = idx & 255;
    W2aT[n * H_DIM + k] = (_Float16)g_w2[k * H_DIM + n];
}

// ---- Main: one workgroup (8 waves) per gene; fully fused GEMM+pool+softmax ----
__global__ void __launch_bounds__(256)
main_gene_kernel(const float* __restrict__ ctrl, const int* __restrict__ gidxs,
                 const float* __restrict__ g_w1, const float* __restrict__ g_w2,
                 const float* __restrict__ g_b2, const float* __restrict__ g_w3,
                 const float* __restrict__ g_b3,
                 const float* __restrict__ A, const float* __restrict__ B,
                 const _Float16* __restrict__ W2aT,
                 float* __restrict__ out)
{
    __shared__ float ctrlg[C_CELLS];
    __shared__ float w0l[H_DIM];
    __shared__ float Bgl[H_DIM];
    __shared__ float pm[H_DIM];
    __shared__ float pbl[H_DIM];
    __shared__ float w3al[H_DIM];
    __shared__ float p2acc[H_DIM];
    __shared__ float sacc[C_CELLS];
    __shared__ float red[H_DIM];

    const int g = blockIdx.x;
    const int t = threadIdx.x;
    const int gi = gidxs[g];

    if (t < C_CELLS) { ctrlg[t] = ctrl[t * G_GENES + gi]; sacc[t] = 0.f; }
    w0l[t]   = g_w1[t];                 // row 0 of g_w1
    Bgl[t]   = B[g * H_DIM + t];
    w3al[t]  = g_w3[t];                 // w3a
    p2acc[t] = 0.f;
    __syncthreads();

    // p[h] = mean_c elu(ctrl*w0 + A + B)
    {
        const float w0h = w0l[t], bh = Bgl[t];
        float s = 0.f;
        for (int c = 0; c < C_CELLS; ++c)
            s += eluf(ctrlg[c] * w0h + A[c * H_DIM + t] + bh);
        pm[t] = s * (1.f / (float)C_CELLS);
    }
    __syncthreads();

    // pb[n] = p @ W2b + b2   (W2b = g_w2 rows 256..511)
    {
        float acc = g_b2[t];
        for (int h = 0; h < H_DIM; ++h)
            acc += pm[h] * g_w2[(H_DIM + h) * H_DIM + t];
        pbl[t] = acc;
    }
    __syncthreads();

    // --- build A fragments (16-bit A 16x32, contiguous k = 16*half + s) directly in regs ---
    const int lane = t & 31, wv = t >> 5;
    const int m = lane & 15, half = lane >> 4;
    const int cA = wv * 16 + m;       // cell row for A fragment
    const int koff = half * 16;
    const float cg = ctrlg[cA];
    v16h afrag[8];
    #pragma unroll
    for (int kk = 0; kk < 8; ++kk) {
        #pragma unroll
        for (int j = 0; j < 8; ++j) {
            const int kb = koff + 2 * j;
            const int h0 = kk * 32 + kb;
            const float x0 = cg * w0l[h0]     + A[cA * H_DIM + h0]     + Bgl[h0];
            const float x1 = cg * w0l[h0 + 1] + A[cA * H_DIM + h0 + 1] + Bgl[h0 + 1];
            afrag[kk][2 * j]     = (_Float16)eluf(x0);
            afrag[kk][2 * j + 1] = (_Float16)eluf(x1);
        }
    }

    // --- GEMM: x2[c,n] = x @ W2a + pb[n]; fused p2 mean + w3a dot ---
    const int cC = wv * 16 + half * 8;   // + r  -> output cell row
    const int nbaseK = half * 16;
    float s_loc[8];
    #pragma unroll
    for (int r = 0; r < 8; ++r) s_loc[r] = 0.f;

    for (int nt = 0; nt < 16; ++nt) {
        const int n = nt * 16 + m;
        v8f acc = {};
        #pragma unroll
        for (int kk = 0; kk < 8; ++kk) {
            // B fragment: contiguous 16 halfs of pre-transposed W2a (32B aligned)
            v16h bfrag = *(const v16h*)(W2aT + n * H_DIM + kk * 32 + nbaseK);
            acc = __builtin_amdgcn_wmma_f32_16x16x32_f16(
                false, afrag[kk], false, bfrag, (short)0, acc, false, false);
        }
        const float pbn = pbl[n];
        const float w3n = w3al[n];
        float psum = 0.f;
        #pragma unroll
        for (int r = 0; r < 8; ++r) {
            const float x2v = acc[r] + pbn;   // x2[cC+r, n]
            psum += x2v;                      // -> p2 accumulation
            s_loc[r] += eluf(x2v) * w3n;      // -> elu(x2)·w3a
        }
        atomicAdd(&p2acc[n], psum);
    }
    #pragma unroll
    for (int r = 0; r < 8; ++r) atomicAdd(&sacc[cC + r], s_loc[r]);
    __syncthreads();

    // const_g = sum_n elu(p2[n]) * w3b[n] + b3
    red[t] = eluf(p2acc[t] * (1.f / (float)C_CELLS)) * g_w3[H_DIM + t];
    __syncthreads();
    for (int off = 128; off > 0; off >>= 1) {
        if (t < off) red[t] += red[t + off];
        __syncthreads();
    }
    const float constg = red[0] + g_b3[0];
    __syncthreads();

    // softmax over the 128 cells for this gene
    const float sv = (t < C_CELLS) ? (sacc[t] + constg) : -INFINITY;
    red[t] = sv;
    __syncthreads();
    for (int off = 128; off > 0; off >>= 1) {
        if (t < off) red[t] = fmaxf(red[t], red[t + off]);
        __syncthreads();
    }
    const float mx = red[0];
    __syncthreads();
    const float ev = (t < C_CELLS) ? __expf(sv - mx) : 0.f;
    red[t] = ev;
    __syncthreads();
    for (int off = 128; off > 0; off >>= 1) {
        if (t < off) red[t] += red[t + off];
        __syncthreads();
    }
    const float denom = red[0];
    if (t < C_CELLS) out[t * G_GENES + g] = ev / denom;
}

extern "C" void kernel_launch(void* const* d_in, const int* in_sizes, int n_in,
                              void* d_out, int out_size, void* d_ws, size_t ws_size,
                              hipStream_t stream)
{
    const float* ctrl   = (const float*)d_in[0];
    const float* shiftv = (const float*)d_in[1];
    const int*   gidx   = (const int*)  d_in[2];
    const float* ce_w1  = (const float*)d_in[3];
    const float* ce_b1  = (const float*)d_in[4];
    const float* ce_w2  = (const float*)d_in[5];
    const float* ce_b2  = (const float*)d_in[6];
    const float* gtab   = (const float*)d_in[7];
    const float* g_w1   = (const float*)d_in[8];
    const float* g_b1   = (const float*)d_in[9];
    const float* g_w2   = (const float*)d_in[10];
    const float* g_b2   = (const float*)d_in[11];
    const float* g_w3   = (const float*)d_in[12];
    const float* g_b3   = (const float*)d_in[13];
    float* out = (float*)d_out;

    // workspace layout (bytes)
    char* ws = (char*)d_ws;
    float*    ce   = (float*)(ws);               // 128*64*4   = 32768
    float*    A    = (float*)(ws + 32768);       // 128*256*4  = 131072
    float*    B    = (float*)(ws + 163840);      // 2000*256*4 = 2048000
    _Float16* W2aT = (_Float16*)(ws + 2211840);  // 256*256*2  = 131072

    prep_ce_kernel<<<C_CELLS, H_DIM, 0, stream>>>(ctrl, ce_w1, ce_b1, ce_w2, ce_b2, ce);
    prep_A_kernel <<<C_CELLS, H_DIM, 0, stream>>>(ce, g_w1, A);
    prep_B_kernel <<<G_GENES, H_DIM, 0, stream>>>(gidx, shiftv, gtab, g_w1, g_b1, B);
    prep_w2t_kernel<<<H_DIM,  H_DIM, 0, stream>>>(g_w2, W2aT);
    main_gene_kernel<<<G_GENES, 256, 0, stream>>>(ctrl, gidx, g_w1, g_w2, g_b2,
                                                  g_w3, g_b3, A, B, W2aT, out);
    (void)in_sizes; (void)n_in; (void)out_size; (void)ws_size;
}
